// VisualCortexLayer_11081015623962
// MI455X (gfx1250) — 8-bit matrix operand order repaired and checked by reading, compiled, not run on hardware
//
#include <hip/hip_runtime.h>
#include <stdint.h>

typedef __attribute__((ext_vector_type(8))) int          v8i;
typedef __attribute__((ext_vector_type(4))) unsigned int v4u;

#define EPSF 1e-5f

__device__ __forceinline__ v8i pack_v8(v4u lo, v4u hi) {
    v8i r;
    r[0] = (int)lo[0]; r[1] = (int)lo[1]; r[2] = (int)lo[2]; r[3] = (int)lo[3];
    r[4] = (int)hi[0]; r[5] = (int)hi[1]; r[6] = (int)hi[2]; r[7] = (int)hi[3];
    return r;
}

__global__ void init_ws_kernel(unsigned* p, int n) {
    for (int i = blockIdx.x * blockDim.x + threadIdx.x; i < n; i += gridDim.x * blockDim.x)
        p[i] = 0u;
}

__global__ void __launch_bounds__(256)
absmax_kernel(const float* __restrict__ src, int n, int bipolar, unsigned* dst) {
    float m = 0.f;
    for (int i = blockIdx.x * blockDim.x + threadIdx.x; i < n; i += gridDim.x * blockDim.x) {
        float v = src[i];
        if (bipolar) v = 2.f * v - 1.f;
        m = fmaxf(m, fabsf(v));
    }
    for (int s = 16; s >= 1; s >>= 1) m = fmaxf(m, __shfl_xor(m, s, 32));
    __shared__ float red[8];
    int lane = threadIdx.x & 31, w = threadIdx.x >> 5;
    if (lane == 0) red[w] = m;
    __syncthreads();
    if (threadIdx.x == 0) {
        float mm = red[0];
        for (int i = 1; i < 8; ++i) mm = fmaxf(mm, red[i]);
        atomicMax(dst, __float_as_uint(mm));
    }
}

__global__ void __launch_bounds__(256)
abssum_kernel(const float* __restrict__ src, int n, float* dst) {
    float s = 0.f;
    for (int i = blockIdx.x * blockDim.x + threadIdx.x; i < n; i += gridDim.x * blockDim.x)
        s += fabsf(src[i]);
    for (int m = 16; m >= 1; m >>= 1) s += __shfl_xor(s, m, 32);
    __shared__ float red[8];
    int lane = threadIdx.x & 31, w = threadIdx.x >> 5;
    if (lane == 0) red[w] = s;
    __syncthreads();
    if (threadIdx.x == 0) {
        float t = 0.f;
        for (int i = 0; i < 8; ++i) t += red[i];
        atomicAdd(dst, t);
    }
}

__global__ void __launch_bounds__(256)
quant_x_kernel(const float* __restrict__ x, const float* __restrict__ scal,
               signed char* __restrict__ xq) {
    __shared__ signed char q[64 * 16];
    const int t = threadIdx.x;
    const int w0 = blockIdx.x * 16, h = blockIdx.y, n = blockIdx.z;
    const float s = 127.f / (scal[0] + EPSF);
    #pragma unroll
    for (int cc = 0; cc < 4; ++cc) {
        int c = cc * 16 + (t >> 4), w = t & 15;
        float v = x[(((n * 64 + c) * 112 + h) * 112) + w0 + w];
        float qq = rintf((2.f * v - 1.f) * s);
        qq = fminf(fmaxf(qq, -128.f), 127.f);
        q[c * 16 + w] = (signed char)(int)qq;
    }
    __syncthreads();
    int w = t >> 4, dw = t & 15;
    unsigned pkd = (unsigned)(unsigned char)q[(dw * 4 + 0) * 16 + w]
                 | ((unsigned)(unsigned char)q[(dw * 4 + 1) * 16 + w] << 8)
                 | ((unsigned)(unsigned char)q[(dw * 4 + 2) * 16 + w] << 16)
                 | ((unsigned)(unsigned char)q[(dw * 4 + 3) * 16 + w] << 24);
    *(unsigned*)(xq + ((((n * 112 + h) * 112) + w0 + w) << 6) + dw * 4) = pkd;
}

__global__ void __launch_bounds__(256)
quant_td_kernel(const float* __restrict__ td, const float* __restrict__ scal,
                signed char* __restrict__ tdq) {
    __shared__ signed char q[128 * 8];
    const int t = threadIdx.x;
    const int w0 = blockIdx.x * 8, h = blockIdx.y, n = blockIdx.z;
    const float s = 127.f / (scal[1] + EPSF);
    #pragma unroll
    for (int cc = 0; cc < 4; ++cc) {
        int c = cc * 32 + (t >> 3), w = t & 7;
        float v = td[(((n * 128 + c) * 56 + h) * 56) + w0 + w];
        float qq = rintf(v * s);
        qq = fminf(fmaxf(qq, -128.f), 127.f);
        q[c * 8 + w] = (signed char)(int)qq;
    }
    __syncthreads();
    int w = t >> 5, dw = t & 31;
    unsigned pkd = (unsigned)(unsigned char)q[(dw * 4 + 0) * 8 + w]
                 | ((unsigned)(unsigned char)q[(dw * 4 + 1) * 8 + w] << 8)
                 | ((unsigned)(unsigned char)q[(dw * 4 + 2) * 8 + w] << 16)
                 | ((unsigned)(unsigned char)q[(dw * 4 + 3) * 8 + w] << 24);
    *(unsigned*)(tdq + (((((n * 56) + h) * 56) + w0 + w) << 7) + dw * 4) = pkd;
}

__device__ __forceinline__ int a_k_of(int g, int j) {
    int V = j >> 2, b = j & 3, vv = V & 3, hi2 = V >> 2;
    return hi2 * 32 + g * 16 + vv * 4 + b;
}

__global__ void pack_w_kernel(const float* __restrict__ cw, const float* __restrict__ scal,
                              signed char* __restrict__ wpack) {
    int tid = blockIdx.x * blockDim.x + threadIdx.x;
    if (tid >= 8 * 9 * 32) return;
    int chunk = tid / (9 * 32), p = (tid / 32) % 9, lane = tid % 32;
    int kh = p / 3, kw = p % 3, g = lane >> 4, m = lane & 15;
    int co = chunk * 16 + m;
    float s = scal[2] * (1.f / 73728.f) + EPSF;
    float invs = 1.f / s;
    #pragma unroll
    for (int j = 0; j < 32; ++j) {
        int ci = a_k_of(g, j);
        float wv = cw[((co * 64 + ci) * 3 + kh) * 3 + kw];
        float qq = fminf(fmaxf(rintf(wv * invs), -1.f), 1.f);
        wpack[tid * 32 + j] = (signed char)(int)qq;
    }
}

__global__ void pack_gw_kernel(const float* __restrict__ gw, const float* __restrict__ scal,
                               signed char* __restrict__ gwpack) {
    int tid = blockIdx.x * blockDim.x + threadIdx.x;
    if (tid >= 8 * 2 * 32) return;
    int chunk = tid / (2 * 32), kc = (tid / 32) % 2, lane = tid % 32;
    int g = lane >> 4, m = lane & 15;
    int co = chunk * 16 + m;
    float s = scal[3] * (1.f / 16384.f) + EPSF;
    float invs = 1.f / s;
    #pragma unroll
    for (int j = 0; j < 32; ++j) {
        int ci = kc * 64 + a_k_of(g, j);
        float wv = gw[co * 128 + ci];
        float qq = fminf(fmaxf(rintf(wv * invs), -1.f), 1.f);
        gwpack[tid * 32 + j] = (signed char)(int)qq;
    }
}

__global__ void __launch_bounds__(256)
conv3x3_wmma_kernel(const signed char* __restrict__ xq,
                    const signed char* __restrict__ wpack,
                    const float* __restrict__ scal,
                    const float* __restrict__ conv_b,
                    float* __restrict__ y,
                    float* __restrict__ stats) {
    __shared__ __align__(16) signed char tile[3 * 18 * 64];
    const int t = threadIdx.x;
    const int w0 = blockIdx.x * 16, h = blockIdx.y, n = blockIdx.z;

    for (int idx = t; idx < 864; idx += 256) {
        int r = idx / 288, rem = idx % 288;
        int j = rem / 16, dwi = rem % 16;
        int hi2 = h - 1 + r, wi2 = w0 - 1 + j;
        unsigned v = 0u;
        if (hi2 >= 0 && hi2 < 112 && wi2 >= 0 && wi2 < 112)
            v = *(const unsigned*)(xq + (((n * 112 + hi2) * 112 + wi2) << 6) + dwi * 4);
        ((unsigned*)tile)[idx] = v;
    }
    if (t == 0 && h < 111)
        __builtin_prefetch(xq + (((n * 112 + h + 1) * 112 + w0) << 6), 0, 1);
    __syncthreads();

    const int wave = t >> 5, lane = t & 31, col = lane & 15, g = lane >> 4;
    v8i acc = {0, 0, 0, 0, 0, 0, 0, 0};
    const signed char* ap = wpack + ((wave * 9) * 32 + lane) * 32;

    #pragma unroll
    for (int kh = 0; kh < 3; ++kh) {
        #pragma unroll
        for (int kw = 0; kw < 3; ++kw) {
            int base = (kh * 18 + col + kw) * 64 + g * 16;
            v4u blo = *(const v4u*)(tile + base);
            v4u bhi = *(const v4u*)(tile + base + 32);
            v4u alo = *(const v4u*)(ap);
            v4u ahi = *(const v4u*)(ap + 16);
            ap += 32 * 32;
            acc = __builtin_amdgcn_wmma_i32_16x16x64_iu8(
                true, pack_v8(alo, ahi), true, pack_v8(blo, bhi), acc, false, false);
        }
    }

    const float maxx = scal[0];
    const float sw = scal[2] * (1.f / 73728.f) + EPSF;
    const float c0 = sw * (maxx + EPSF) * (1.f / 127.f);
    const int slot = (blockIdx.x + 7 * (blockIdx.y + 112 * blockIdx.z)) & 63;
    #pragma unroll
    for (int r = 0; r < 8; ++r) {
        int c = wave * 16 + r + g * 8;
        float yv = (float)acc[r] * c0 + conv_b[c];
        y[(((n * 128 + c) * 112 + h) * 112) + w0 + col] = yv;
        float s = yv, sq = yv * yv;
        #pragma unroll
        for (int m = 8; m >= 1; m >>= 1) {
            s  += __shfl_xor(s,  m, 32);
            sq += __shfl_xor(sq, m, 32);
        }
        if (col == 0) {
            atomicAdd(&stats[((slot * 128) + c) * 2 + 0], s);
            atomicAdd(&stats[((slot * 128) + c) * 2 + 1], sq);
        }
    }
}

__global__ void __launch_bounds__(256)
gate1x1_wmma_kernel(const signed char* __restrict__ tdq,
                    const signed char* __restrict__ gwpack,
                    const float* __restrict__ scal,
                    const float* __restrict__ gate_b,
                    float* __restrict__ gate) {
    const int t = threadIdx.x, wave = t >> 5, lane = t & 31, col = lane & 15, g = lane >> 4;
    const int pos0 = blockIdx.x * 16;
    const signed char* bp = tdq + (size_t)(pos0 + col) * 128 + g * 16;
    v8i acc = {0, 0, 0, 0, 0, 0, 0, 0};
    #pragma unroll
    for (int kc = 0; kc < 2; ++kc) {
        v4u blo = *(const v4u*)(bp + kc * 64);
        v4u bhi = *(const v4u*)(bp + kc * 64 + 32);
        const signed char* ap = gwpack + (((wave * 2 + kc) * 32 + lane) * 32);
        v4u alo = *(const v4u*)ap;
        v4u ahi = *(const v4u*)(ap + 16);
        acc = __builtin_amdgcn_wmma_i32_16x16x64_iu8(
            true, pack_v8(alo, ahi), true, pack_v8(blo, bhi), acc, false, false);
    }
    const float maxtd = scal[1];
    const float sg = scal[3] * (1.f / 16384.f) + EPSF;
    const float c1 = sg * (maxtd + EPSF) * (1.f / 127.f);
    const int pos = pos0 + col, nb = pos / 3136, hw = pos % 3136;
    #pragma unroll
    for (int r = 0; r < 8; ++r) {
        int c = wave * 16 + r + g * 8;
        float lg = (float)acc[r] * c1 + gate_b[c];
        gate[((nb * 128 + c) * 3136) + hw] = 1.f / (1.f + __expf(-lg));
    }
}

__global__ void bn_combine_kernel(const float* __restrict__ stats,
                                  const float* __restrict__ gamma,
                                  const float* __restrict__ beta,
                                  float* __restrict__ bn_ab) {
    int c = threadIdx.x;
    float s = 0.f, sq = 0.f;
    for (int k = 0; k < 64; ++k) {
        s  += stats[((k * 128) + c) * 2 + 0];
        sq += stats[((k * 128) + c) * 2 + 1];
    }
    const float cnt = 16.f * 112.f * 112.f;
    float mu = s / cnt;
    float var = sq / cnt - mu * mu;
    float a = gamma[c] * rsqrtf(var + EPSF);
    bn_ab[c * 2 + 0] = a;
    bn_ab[c * 2 + 1] = beta[c] - mu * a;
}

__global__ void __launch_bounds__(256)
finalize_kernel(const float* __restrict__ y, const float* __restrict__ bn_ab,
                const float* __restrict__ gate, float* __restrict__ out) {
    int idx = blockIdx.x * 256 + threadIdx.x;
    int hw = idx % 12544;
    int nc = idx / 12544;
    int c = nc % 128, nb = nc / 128;
    int h = hw / 112, w = hw % 112;
    float mem = y[idx] * bn_ab[c * 2] + bn_ab[c * 2 + 1];
    float gv = gate[((nb * 128 + c) * 3136) + (h >> 1) * 56 + (w >> 1)];
    mem *= (1.f + gv);
    out[idx] = (mem > 1.0f) ? 1.f : 0.f;
}

extern "C" void kernel_launch(void* const* d_in, const int* in_sizes, int n_in,
                              void* d_out, int out_size, void* d_ws, size_t ws_size,
                              hipStream_t stream) {
    const float* x     = (const float*)d_in[0];
    const float* td    = (const float*)d_in[1];
    const float* cw    = (const float*)d_in[2];
    const float* cb    = (const float*)d_in[3];
    const float* gamma = (const float*)d_in[4];
    const float* beta  = (const float*)d_in[5];
    const float* gw    = (const float*)d_in[6];
    const float* gb    = (const float*)d_in[7];
    (void)in_sizes; (void)n_in; (void)out_size; (void)ws_size;

    char* ws = (char*)d_ws;
    float*       scal   = (float*)ws;
    float*       stats  = (float*)(ws + 64);
    float*       bn_ab  = (float*)(ws + 65600);
    signed char* wpack  = (signed char*)(ws + 66624);
    signed char* gwpack = (signed char*)(ws + 140352);
    signed char* xq     = (signed char*)(ws + 156736);
    signed char* tdq    = (signed char*)(ws + 13001792);
    float*       y      = (float*)(ws + 19424320);
    float*       gate   = (float*)(ws + 122184768);
    float*       out    = (float*)d_out;

    init_ws_kernel<<<65, 256, 0, stream>>>((unsigned*)ws, 16400);

    absmax_kernel<<<1024, 256, 0, stream>>>(x, 16 * 64 * 112 * 112, 1, (unsigned*)&scal[0]);
    absmax_kernel<<<512, 256, 0, stream>>>(td, 16 * 128 * 56 * 56, 0, (unsigned*)&scal[1]);
    abssum_kernel<<<36, 256, 0, stream>>>(cw, 73728, &scal[2]);
    abssum_kernel<<<8, 256, 0, stream>>>(gw, 16384, &scal[3]);

    quant_x_kernel<<<dim3(7, 112, 16), 256, 0, stream>>>(x, scal, xq);
    quant_td_kernel<<<dim3(7, 56, 16), 256, 0, stream>>>(td, scal, tdq);
    pack_w_kernel<<<9, 256, 0, stream>>>(cw, scal, wpack);
    pack_gw_kernel<<<2, 256, 0, stream>>>(gw, scal, gwpack);

    conv3x3_wmma_kernel<<<dim3(7, 112, 16), 256, 0, stream>>>(xq, wpack, scal, cb, y, stats);
    gate1x1_wmma_kernel<<<3136, 256, 0, stream>>>(tdq, gwpack, scal, gb, gate);

    bn_combine_kernel<<<1, 128, 0, stream>>>(stats, gamma, beta, bn_ab);
    finalize_kernel<<<100352, 256, 0, stream>>>(y, bn_ab, gate, out);
}
